// GPNA_12429635355187
// MI455X (gfx1250) — matrix operand order repaired and checked by reading, not compiled, not run on hardware
//
#include <hip/hip_runtime.h>
#include <stdint.h>

#define NN 50000
#define EE 500000

typedef __attribute__((ext_vector_type(16))) __bf16 v16bf;
typedef __attribute__((ext_vector_type(8)))  float  v8f;
typedef __attribute__((ext_vector_type(4)))  float  v4f;
typedef unsigned short bfr;   // raw bf16 storage

union AFrag { v16bf v; v4f f4[2]; };

__device__ __forceinline__ bfr f2b(float f) {
  unsigned u = __float_as_uint(f);
  unsigned r = (u + 0x7FFFu + ((u >> 16) & 1u)) >> 16;   // RNE
  return (bfr)r;
}

__device__ __forceinline__ void atomicMaxF(float* a, float v) {
  if (v >= 0.f) atomicMax((int*)a, __float_as_int(v));
  else          atomicMin((unsigned int*)a, (unsigned int)__float_as_int(v));
}
__device__ __forceinline__ void atomicMinF(float* a, float v) {
  if (v >= 0.f) atomicMin((int*)a, __float_as_int(v));
  else          atomicMax((unsigned int*)a, (unsigned int)__float_as_int(v));
}

// ---------------- small elementwise / reduction kernels ----------------

__global__ void k_deg(const int* __restrict__ src, const int* __restrict__ dst,
                      float* __restrict__ deg, float* __restrict__ degall) {
  int e = blockIdx.x * blockDim.x + threadIdx.x;
  if (e >= EE) return;
  atomicAdd(&deg[dst[e]], 1.f);
  atomicAdd(&degall[src[e]], 1.f);
  atomicAdd(&degall[dst[e]], 1.f);
}

__global__ void k_sumlog(const float* __restrict__ degall, float* __restrict__ avgsum) {
  __shared__ float sh[256];
  float s = 0.f;
  for (int i = blockIdx.x * 256 + threadIdx.x; i < NN; i += gridDim.x * 256)
    s += logf(degall[i] + 1.f);
  sh[threadIdx.x] = s; __syncthreads();
  for (int o = 128; o > 0; o >>= 1) {
    if (threadIdx.x < o) sh[threadIdx.x] += sh[threadIdx.x + o];
    __syncthreads();
  }
  if (threadIdx.x == 0) atomicAdd(avgsum, sh[0]);
}

__global__ void k_add(const float* __restrict__ a, const float* __restrict__ b,
                      float* __restrict__ o, int n) {
  int i = blockIdx.x * blockDim.x + threadIdx.x;
  if (i < n) o[i] = a[i] + b[i];
}

__global__ void k_cvt(const float* __restrict__ in, bfr* __restrict__ out, int n) {
  int i = blockIdx.x * blockDim.x + threadIdx.x;
  if (i < n) out[i] = f2b(in[i]);
}

// W[K,N] f32  ->  Wt[N,K] bf16 (transposed for contiguous-K B fragments)
__global__ void k_cvtT(const float* __restrict__ W, bfr* __restrict__ Wt, int K, int N) {
  int idx = blockIdx.x * blockDim.x + threadIdx.x;
  if (idx >= K * N) return;
  int n = idx / K, k = idx % K;
  Wt[idx] = f2b(W[(size_t)k * N + n]);
}

__global__ void k_init_agg(float* s, float* q, float* mn, float* mx, int n) {
  int i = blockIdx.x * blockDim.x + threadIdx.x;
  if (i >= n) return;
  s[i] = 0.f; q[i] = 0.f; mn[i] = 3.402823466e+38f; mx[i] = -3.402823466e+38f;
}

// Build edge operand matrix  A_pre[e] = bf16(cat(h[dst[e]], h[src[e]]))
__global__ void k_gather(const float* __restrict__ hin, const int* __restrict__ src,
                         const int* __restrict__ dst, bfr* __restrict__ Apre, int F) {
  long long idx = (long long)blockIdx.x * blockDim.x + threadIdx.x;
  int perRow = (2 * F) >> 2;
  long long tot = (long long)EE * perRow;
  if (idx >= tot) return;
  int e = (int)(idx / perRow);
  int c = (int)(idx % perRow) * 4;
  int node, cc;
  if (c < F) { node = dst[e]; cc = c; } else { node = src[e]; cc = c - F; }
  v4f v = *(const v4f*)(hin + (size_t)node * F + cc);
  bfr* o = Apre + (size_t)e * 2 * F + c;
  o[0] = f2b(v.x); o[1] = f2b(v.y); o[2] = f2b(v.z); o[3] = f2b(v.w);
}

// agg buffers + degrees -> bf16 node matrix [x | agg | agg*s1 | agg*s2]  (13F cols)
__global__ void k_finalize(const float* __restrict__ hin, const float* __restrict__ asum,
                           const float* __restrict__ asq, const float* __restrict__ amn,
                           const float* __restrict__ amx, const float* __restrict__ deg,
                           const float* __restrict__ avgsum, bfr* __restrict__ Apost, int F) {
  int idx = blockIdx.x * blockDim.x + threadIdx.x;
  if (idx >= NN * F) return;
  int n = idx / F, c = idx % F;
  float d    = deg[n];
  float degc = fmaxf(d, 1.f);
  float inv  = 1.f / degc;
  float mean = asum[idx] * inv;
  float m2   = asq[idx] * inv;
  float stdv = sqrtf(fmaxf(m2 - mean * mean, 0.f) + 1e-5f);
  bool  has  = d > 0.f;
  float mn   = has ? amn[idx] : 0.f;
  float mx   = has ? amx[idx] : 0.f;
  float avg  = avgsum[0] / (float)NN;
  float logd = logf(degc + 1.f);
  float s1 = logd / avg, s2 = avg / logd;
  bfr* row = Apost + (size_t)n * 13 * F;
  row[c]        = f2b(hin[(size_t)n * F + c]);
  row[F + c]    = f2b(mean);        row[2*F + c] = f2b(mn);
  row[3*F + c]  = f2b(mx);          row[4*F + c] = f2b(stdv);
  row[5*F + c]  = f2b(mean * s1);   row[6*F + c] = f2b(mn * s1);
  row[7*F + c]  = f2b(mx * s1);     row[8*F + c] = f2b(stdv * s1);
  row[9*F + c]  = f2b(mean * s2);   row[10*F + c] = f2b(mn * s2);
  row[11*F + c] = f2b(mx * s2);     row[12*F + c] = f2b(stdv * s2);
}

__global__ void k_bn_stats(const float* __restrict__ X, float* __restrict__ mu,
                           float* __restrict__ rvar, int M, int Ncols) {
  int c = blockIdx.x;
  float s = 0.f, q = 0.f;
  for (int r = threadIdx.x; r < M; r += 256) {
    float v = X[(size_t)r * Ncols + c];
    s += v; q += v * v;
  }
  __shared__ float ss[256], qq[256];
  ss[threadIdx.x] = s; qq[threadIdx.x] = q; __syncthreads();
  for (int o = 128; o > 0; o >>= 1) {
    if (threadIdx.x < o) { ss[threadIdx.x] += ss[threadIdx.x+o]; qq[threadIdx.x] += qq[threadIdx.x+o]; }
    __syncthreads();
  }
  if (threadIdx.x == 0) {
    float m = ss[0] / M;
    mu[c] = m;
    rvar[c] = rsqrtf(qq[0] / M - m * m + 1e-5f);
  }
}

__global__ void k_bn_elu(const float* __restrict__ X, const float* __restrict__ mu,
                         const float* __restrict__ rvar, const float* __restrict__ gamma,
                         const float* __restrict__ beta, float* __restrict__ H,
                         float* __restrict__ B4, int total, int Ncols) {
  int idx = blockIdx.x * blockDim.x + threadIdx.x;
  if (idx >= total) return;
  int c = idx % Ncols;
  float y = gamma[c] * (X[idx] - mu[c]) * rvar[c] + beta[c];
  if (B4) B4[idx] = y;
  H[idx] = y > 0.f ? y : (expf(y) - 1.f);
}

__global__ void k_cls(const float* __restrict__ h4, const float* __restrict__ Wc,
                      const float* __restrict__ bc, float* __restrict__ out) {
  int n = blockIdx.x * blockDim.x + threadIdx.x;
  if (n >= NN) return;
  float a0 = bc[0], a1 = bc[1];
  const float* r = h4 + (size_t)n * 64;
  #pragma unroll 8
  for (int k = 0; k < 64; k++) { float v = r[k]; a0 += v * Wc[2*k]; a1 += v * Wc[2*k + 1]; }
  out[2*n] = a0; out[2*n + 1] = a1;
}

// ---------------- WMMA tiled GEMM (bf16 -> f32), optional fused segment-agg ----------------
// C[M,N] = A[M,K](bf16) x Bt[N,K]^T(bf16) + bias.  Block = 256 thr (8 waves), tile 128 x N.
// EDGE=true: instead of storing C, do L2-resident atomic sum/sumsq/min/max keyed by dst[row].
// Inner loop is software-pipelined: B fragments double-buffered so ds_load_b128 of the next
// fragment overlaps the current v_wmma (avoids s_wait_dscnt 0 before every WMMA).

#define LDS_STRIDE 72   // 64 halves + 8 pad: distinct banks across 16 lanes, 16B-aligned

template<int NTC, bool EDGE>
__global__ __launch_bounds__(256)
void gemm_k(const bfr* __restrict__ A, const bfr* __restrict__ Bt,
            const float* __restrict__ bias, float* __restrict__ C, int M, int K,
            const int* __restrict__ dst, float* __restrict__ asum, float* __restrict__ asq,
            float* __restrict__ amn, float* __restrict__ amx) {
  constexpr int NB = NTC * 16;
  extern __shared__ char smem[];
  bfr* lAs = (bfr*)smem;              // 128 x 64 (stride 72)
  bfr* lBs = lAs + 128 * LDS_STRIDE;  // NB  x 64 (stride 72)

  const int tid  = threadIdx.x;
  const int wave = tid >> 5, lane = tid & 31;
  const int wrow = wave * 16;
  const int mb   = blockIdx.x * 128;

  v8f zero = {0.f,0.f,0.f,0.f,0.f,0.f,0.f,0.f};
  v8f acc[NTC];
  #pragma unroll
  for (int i = 0; i < NTC; i++) acc[i] = zero;

  // per-lane fragment base offsets
  const int a_row = (wrow + (lane & 15)) * LDS_STRIDE;
  const int a_kh  = (lane >> 4) * 8;
  const int b_col = (lane & 15) * LDS_STRIDE;
  const int b_kh  = (lane >> 4) * 8;

  for (int kb = 0; kb < K; kb += 64) {
    // stage A slab: 128 rows x 64 halves (zero-pad rows >= M)
    #pragma unroll
    for (int i = 0; i < 4; i++) {
      int chunk = tid + i * 256;        // 1024 chunks of 8 halves
      int r = chunk >> 3, p = chunk & 7;
      v4f val = {0.f,0.f,0.f,0.f};
      if (mb + r < M)
        val = *(const v4f*)(A + (size_t)(mb + r) * K + kb + p * 8);
      *(v4f*)(lAs + r * LDS_STRIDE + p * 8) = val;
    }
    // stage B slab: NB rows (cols of W) x 64 halves
    for (int chunk = tid; chunk < NB * 8; chunk += 256) {
      int n = chunk >> 3, p = chunk & 7;
      *(v4f*)(lBs + n * LDS_STRIDE + p * 8) =
        *(const v4f*)(Bt + (size_t)n * K + kb + p * 8);
    }
    if (kb + 64 < K && mb + (tid >> 1) < M)
      __builtin_prefetch(A + (size_t)(mb + (tid >> 1)) * K + kb + 64, 0, 1);
    __syncthreads();

    // hoist both A fragments (k0 = 0 and 32)
    AFrag a0, a1;
    a0.f4[0] = *(const v4f*)(lAs + a_row + 0  + a_kh);
    a0.f4[1] = *(const v4f*)(lAs + a_row + 16 + a_kh);
    a1.f4[0] = *(const v4f*)(lAs + a_row + 32 + a_kh);
    a1.f4[1] = *(const v4f*)(lAs + a_row + 48 + a_kh);

    // flattened (k0, nt) WMMA chain with 2-deep B double-buffer
    AFrag bf[2];
    {
      int bo = b_col + 0 + b_kh;            // idx 0: k0=0, nt=0
      bf[0].f4[0] = *(const v4f*)(lBs + bo);
      bf[0].f4[1] = *(const v4f*)(lBs + bo + 16);
    }
    #pragma unroll
    for (int idx = 0; idx < 2 * NTC; idx++) {
      if (idx + 1 < 2 * NTC) {
        int nidx = idx + 1;
        int k0n  = (nidx >= NTC) ? 32 : 0;
        int ntn  = (nidx >= NTC) ? nidx - NTC : nidx;
        int bo = ntn * 16 * LDS_STRIDE + b_col + k0n + b_kh;
        bf[(idx + 1) & 1].f4[0] = *(const v4f*)(lBs + bo);
        bf[(idx + 1) & 1].f4[1] = *(const v4f*)(lBs + bo + 16);
      }
      int nt = (idx >= NTC) ? idx - NTC : idx;
      const AFrag& aa = (idx >= NTC) ? a1 : a0;
      acc[nt] = __builtin_amdgcn_wmma_f32_16x16x32_bf16(
          false, aa.v, false, bf[idx & 1].v, (short)0, acc[nt], false, false);
    }
    __syncthreads();
  }

  // epilogue: C lane layout -> VGPR i: M = i + (lane>=16 ? 8 : 0), N = lane&15
  const int colbase = lane & 15;
  const int rbase   = mb + wrow + ((lane >> 4) ? 8 : 0);
  #pragma unroll
  for (int nt = 0; nt < NTC; nt++) {
    int col = nt * 16 + colbase;
    float bv = bias[col];
    #pragma unroll
    for (int i = 0; i < 8; i++) {
      int r = rbase + i;
      if (r < M) {
        float v = acc[nt][i] + bv;
        if constexpr (EDGE) {
          int d = dst[r];
          size_t off = (size_t)d * NB + col;
          atomicAdd(asum + off, v);
          atomicAdd(asq  + off, v * v);
          atomicMinF(amn + off, v);
          atomicMaxF(amx + off, v);
        } else {
          C[(size_t)r * NB + col] = v;
        }
      }
    }
  }
}

static void launch_gemm_rows(int NTC, const bfr* A, const bfr* Bt, const float* bias,
                             float* C, int M, int K, hipStream_t st) {
  dim3 g((M + 127) / 128), b(256);
  size_t sm = (size_t)(128 * LDS_STRIDE + NTC * 16 * LDS_STRIDE) * 2;
  switch (NTC) {
    case 4:  gemm_k<4,  false><<<g, b, sm, st>>>(A, Bt, bias, C, M, K, nullptr, nullptr, nullptr, nullptr, nullptr); break;
    case 8:  gemm_k<8,  false><<<g, b, sm, st>>>(A, Bt, bias, C, M, K, nullptr, nullptr, nullptr, nullptr, nullptr); break;
    case 16: gemm_k<16, false><<<g, b, sm, st>>>(A, Bt, bias, C, M, K, nullptr, nullptr, nullptr, nullptr, nullptr); break;
  }
}

static void launch_gemm_edge(int NTC, const bfr* A, const bfr* Bt, const float* bias,
                             int M, int K, const int* dst, float* s, float* q,
                             float* mn, float* mx, hipStream_t st) {
  dim3 g((M + 127) / 128), b(256);
  size_t sm = (size_t)(128 * LDS_STRIDE + NTC * 16 * LDS_STRIDE) * 2;
  switch (NTC) {
    case 8:  gemm_k<8,  true><<<g, b, sm, st>>>(A, Bt, bias, nullptr, M, K, dst, s, q, mn, mx); break;
    case 16: gemm_k<16, true><<<g, b, sm, st>>>(A, Bt, bias, nullptr, M, K, dst, s, q, mn, mx); break;
  }
}

// ---------------- host orchestration ----------------

extern "C" void kernel_launch(void* const* d_in, const int* in_sizes, int n_in,
                              void* d_out, int out_size, void* d_ws, size_t ws_size,
                              hipStream_t stream) {
  const float* x  = (const float*)d_in[0];
  const int*   ei = (const int*)d_in[1];
  const int* src = ei;
  const int* dst = ei + EE;
  const float* Wc = (const float*)d_in[34];
  const float* bc = (const float*)d_in[35];

  char* p = (char*)d_ws;
  auto alloc = [&](size_t bytes) -> void* {
    void* r = (void*)p; p += (bytes + 255) & ~(size_t)255; return r;
  };

  float* deg    = (float*)alloc((size_t)NN * 4);
  float* degall = (float*)alloc((size_t)NN * 4);
  float* avgsum = (float*)alloc(256);
  float* h1     = (float*)alloc((size_t)NN * 128 * 4);
  float* h2     = (float*)alloc((size_t)NN * 256 * 4);
  float* h3     = (float*)alloc((size_t)NN * 128 * 4);
  float* hin4   = (float*)alloc((size_t)NN * 128 * 4);
  float* h4     = (float*)alloc((size_t)NN * 64 * 4);
  bfr*   Apre   = (bfr*)  alloc((size_t)EE * 512 * 2);
  float* asum   = (float*)alloc((size_t)NN * 256 * 4);
  float* asq    = (float*)alloc((size_t)NN * 256 * 4);
  float* amn    = (float*)alloc((size_t)NN * 256 * 4);
  float* amx    = (float*)alloc((size_t)NN * 256 * 4);
  bfr*   Apost  = (bfr*)  alloc((size_t)NN * 3328 * 2);
  float* post   = (float*)alloc((size_t)NN * 256 * 4);
  bfr*   postbf = (bfr*)  alloc((size_t)NN * 256 * 2);
  float* linbuf = (float*)alloc((size_t)NN * 256 * 4);
  bfr*   wpreT  = (bfr*)  alloc((size_t)256 * 512 * 2);
  bfr*   wpostT = (bfr*)  alloc((size_t)3328 * 256 * 2);
  bfr*   wlinT  = (bfr*)  alloc((size_t)256 * 256 * 2);
  float* mu     = (float*)alloc(1024);
  float* rvar   = (float*)alloc(1024);

  hipMemsetAsync(deg,    0, (size_t)NN * 4, stream);
  hipMemsetAsync(degall, 0, (size_t)NN * 4, stream);
  hipMemsetAsync(avgsum, 0, 4, stream);
  k_deg<<<(EE + 255) / 256, 256, 0, stream>>>(src, dst, deg, degall);
  k_sumlog<<<64, 256, 0, stream>>>(degall, avgsum);

  float* outp = (float*)d_out;
  float* p4   = outp + (size_t)NN * 2;
  float* b4   = p4 + (size_t)NN * 64;

  const int Fs[4]  = {128, 128, 256, 128};
  const int Fos[4] = {128, 256, 128, 64};
  const float* hins[4]  = {x, h1, h2, hin4};
  float*       houts[4] = {h1, h2, h3, h4};

  for (int l = 0; l < 4; l++) {
    int F = Fs[l], Fo = Fos[l], K1 = 2 * F, K2 = 13 * F;
    const float* Wpre  = (const float*)d_in[2 + l*8 + 0];
    const float* bpre  = (const float*)d_in[2 + l*8 + 1];
    const float* Wpost = (const float*)d_in[2 + l*8 + 2];
    const float* bpost = (const float*)d_in[2 + l*8 + 3];
    const float* Wlin  = (const float*)d_in[2 + l*8 + 4];
    const float* blin  = (const float*)d_in[2 + l*8 + 5];
    const float* gamma = (const float*)d_in[2 + l*8 + 6];
    const float* beta  = (const float*)d_in[2 + l*8 + 7];

    if (l == 3)
      k_add<<<((size_t)NN * 128 + 255) / 256, 256, 0, stream>>>(h3, h1, hin4, NN * 128);

    k_cvtT<<<(K1 * F  + 255) / 256, 256, 0, stream>>>(Wpre,  wpreT,  K1, F);
    k_cvtT<<<(K2 * Fo + 255) / 256, 256, 0, stream>>>(Wpost, wpostT, K2, Fo);
    k_cvtT<<<(Fo * Fo + 255) / 256, 256, 0, stream>>>(Wlin,  wlinT,  Fo, Fo);
    k_init_agg<<<(NN * F + 255) / 256, 256, 0, stream>>>(asum, asq, amn, amx, NN * F);

    long long gtot = (long long)EE * (2 * F / 4);
    k_gather<<<(unsigned)((gtot + 255) / 256), 256, 0, stream>>>(hins[l], src, dst, Apre, F);

    launch_gemm_edge(F / 16, Apre, wpreT, bpre, EE, K1, dst, asum, asq, amn, amx, stream);

    k_finalize<<<(NN * F + 255) / 256, 256, 0, stream>>>(hins[l], asum, asq, amn, amx,
                                                         deg, avgsum, Apost, F);

    launch_gemm_rows(Fo / 16, Apost, wpostT, bpost, post, NN, K2, stream);
    k_cvt<<<(NN * Fo + 255) / 256, 256, 0, stream>>>(post, postbf, NN * Fo);

    float* linC = (l == 3) ? p4 : linbuf;
    launch_gemm_rows(Fo / 16, postbf, wlinT, blin, linC, NN, Fo, stream);

    k_bn_stats<<<Fo, 256, 0, stream>>>(linC, mu, rvar, NN, Fo);
    k_bn_elu<<<(NN * Fo + 255) / 256, 256, 0, stream>>>(linC, mu, rvar, gamma, beta,
                                                        houts[l], (l == 3) ? b4 : nullptr,
                                                        NN * Fo, Fo);
  }

  k_cls<<<(NN + 255) / 256, 256, 0, stream>>>(h4, Wc, bc, outp);
}
